// TDMPC2Policy_70128226009378
// MI455X (gfx1250) — matrix operand order repaired and checked by reading, not compiled, not run on hardware
//
#include <hip/hip_runtime.h>
#include <stddef.h>

// ---------------------------------------------------------------------------
// Types for CDNA5 WMMA
// ---------------------------------------------------------------------------
typedef __attribute__((ext_vector_type(16))) _Float16 v16h;
typedef __attribute__((ext_vector_type(8)))  _Float16 v8h;   // 16-byte load unit
typedef __attribute__((ext_vector_type(8)))  float    v8f;

#define DEV_INLINE __device__ __forceinline__

// ---------------------------------------------------------------------------
// RNG helpers (deterministic, counter based)
// ---------------------------------------------------------------------------
DEV_INLINE unsigned long long sm64(unsigned long long x) {
  x += 0x9E3779B97F4A7C15ull;
  x = (x ^ (x >> 30)) * 0xBF58476D1CE4E5B9ull;
  x = (x ^ (x >> 27)) * 0x94D049BB133111EBull;
  return x ^ (x >> 31);
}
DEV_INLINE float u01(unsigned long long h) {
  return ((float)((h >> 40) & 0xFFFFFFull) + 0.5f) * (1.0f / 16777216.0f);
}
DEV_INLINE float nrand(unsigned long long seed, unsigned long long idx) {
  unsigned long long h1 = sm64(seed * 0x100000001B3ull + idx * 2u + 1u);
  unsigned long long h2 = sm64(h1 + 0xDA3E39CB94B95BDBull);
  float r = sqrtf(-2.0f * logf(u01(h1)));
  return r * cosf(6.28318530718f * u01(h2));
}

// ---------------------------------------------------------------------------
// Fragment loaders (ISA 7.12.2 layouts)
// ---------------------------------------------------------------------------
DEV_INLINE v16h load_fragA(const _Float16* p) {   // 16x32 A: VGPR0-3 k+0..7, VGPR4-7 k+16..23
  v8h lo = *(const v8h*)(p);
  v8h hi = *(const v8h*)(p + 16);
  return __builtin_shufflevector(lo, hi, 0, 1, 2, 3, 4, 5, 6, 7,
                                 8, 9, 10, 11, 12, 13, 14, 15);
}
DEV_INLINE v16h load_fragB(const _Float16* p) {   // 32x16 B: VGPR0-3 k+0..7, VGPR4-7 k+16..23
  v8h lo = *(const v8h*)(p);
  v8h hi = *(const v8h*)(p + 16);
  return __builtin_shufflevector(lo, hi, 0, 1, 2, 3, 4, 5, 6, 7,
                                 8, 9, 10, 11, 12, 13, 14, 15);
}

// ---------------------------------------------------------------------------
// Fused GEMM + epilogue kernel.
//   Y = epilogue(X @ W^T + bias)
//   X : M x K       fp16, row-major, ld = ldx (K multiple of 32, rows 16B aligned)
//   W : (NW*64) x K fp16, row-major, ld = ldw (so B = W^T is K x N)
//   Block = 32 rows x (NW*64) cols, NW waves; each wave: 2 M-tiles x 4 N-tiles,
//   software-pipelined (double-buffered fragments), wave32 WMMA 16x16x32.
//   MODE 0: LayerNorm + mish            -> fp16 Yh
//   MODE 1: LayerNorm + simnorm (8-grp) -> fp16 Yh
//   MODE 2: bias only (head)            -> fp32 Yf, cols < OUT
// ---------------------------------------------------------------------------
template <int NW, int MODE>
__global__ void __launch_bounds__(NW * 32)
gemm_ln_act(const _Float16* __restrict__ X, int ldx,
            const _Float16* __restrict__ W, int ldw,
            const float* __restrict__ bias,
            const float* __restrict__ gamma,
            const float* __restrict__ beta,
            _Float16* __restrict__ Yh, int ldyh,
            float* __restrict__ Yf, int ldyf,
            int K, int OUT) {
  constexpr int C = NW * 64;                  // columns per block
  __shared__ float part_s[32 * NW];
  __shared__ float part_q[32 * NW];
  __shared__ float rmean[32];
  __shared__ float rrstd[32];

  const int lane = threadIdx.x & 31;
  const int wave = threadIdx.x >> 5;
  const int row0 = blockIdx.x * 32;
  const int ln15 = lane & 15;
  const int hi16 = lane >> 4;                 // 0 | 1
  const int kg8  = hi16 * 8;                  // A-fragment k sub-offset
  const int kg16 = hi16 * 16;                 // unused; B uses kg8 as well
  const int base = wave * 64;

  // Fragment base pointers
  const _Float16* pA[2];
#pragma unroll
  for (int m = 0; m < 2; ++m)
    pA[m] = X + (size_t)(row0 + m * 16 + ln15) * ldx + kg8;
  const _Float16* pB[4];
#pragma unroll
  for (int t = 0; t < 4; ++t)
    pB[t] = W + (size_t)(base + t * 16 + ln15) * ldw + kg8;

  v8f acc[2][4];
#pragma unroll
  for (int m = 0; m < 2; ++m)
#pragma unroll
    for (int t = 0; t < 4; ++t)
      acc[m][t] = (v8f){0.f, 0.f, 0.f, 0.f, 0.f, 0.f, 0.f, 0.f};

  // --- software-pipelined main loop (ping-pong buffers, no reg rotation) ---
  v16h aP[2], bP[4], aQ[2], bQ[4];
#pragma unroll
  for (int m = 0; m < 2; ++m) aP[m] = load_fragA(pA[m]);
#pragma unroll
  for (int t = 0; t < 4; ++t) bP[t] = load_fragB(pB[t]);

  int k0 = 0;
  while (true) {
    int k1 = k0 + 32;
    if (k1 < K) {
#pragma unroll
      for (int m = 0; m < 2; ++m) aQ[m] = load_fragA(pA[m] + k1);
#pragma unroll
      for (int t = 0; t < 4; ++t) bQ[t] = load_fragB(pB[t] + k1);
      __builtin_prefetch(pA[0] + k1 + 32, 0, 1);
    }
#pragma unroll
    for (int m = 0; m < 2; ++m)
#pragma unroll
      for (int t = 0; t < 4; ++t)
        acc[m][t] = __builtin_amdgcn_wmma_f32_16x16x32_f16(
            false, aP[m], false, bP[t], (short)0, acc[m][t], false, false);
    if (k1 >= K) break;

    int k2 = k1 + 32;
    if (k2 < K) {
#pragma unroll
      for (int m = 0; m < 2; ++m) aP[m] = load_fragA(pA[m] + k2);
#pragma unroll
      for (int t = 0; t < 4; ++t) bP[t] = load_fragB(pB[t] + k2);
      __builtin_prefetch(pA[0] + k2 + 32, 0, 1);
    }
#pragma unroll
    for (int m = 0; m < 2; ++m)
#pragma unroll
      for (int t = 0; t < 4; ++t)
        acc[m][t] = __builtin_amdgcn_wmma_f32_16x16x32_f16(
            false, aQ[m], false, bQ[t], (short)0, acc[m][t], false, false);
    if (k2 >= K) break;
    k0 = k2;
  }

  // --- add bias (pre-LN affine input is xW^T + b) ---
  float biasv[4];
#pragma unroll
  for (int t = 0; t < 4; ++t) {
    int c = base + t * 16 + ln15;
    biasv[t] = (c < OUT) ? bias[c] : 0.f;
  }
#pragma unroll
  for (int m = 0; m < 2; ++m)
#pragma unroll
    for (int t = 0; t < 4; ++t)
#pragma unroll
      for (int v = 0; v < 8; ++v)
        acc[m][t][v] += biasv[t];

  if (MODE != 2) {
    // ---- per-row stats: butterfly within 16-lane halves; rows differ per half
#pragma unroll
    for (int m = 0; m < 2; ++m) {
#pragma unroll
      for (int v = 0; v < 8; ++v) {
        float s = acc[m][0][v] + acc[m][1][v] + acc[m][2][v] + acc[m][3][v];
        float q = acc[m][0][v] * acc[m][0][v] + acc[m][1][v] * acc[m][1][v] +
                  acc[m][2][v] * acc[m][2][v] + acc[m][3][v] * acc[m][3][v];
#pragma unroll
        for (int mk = 1; mk <= 8; mk <<= 1) {
          s += __shfl_xor(s, mk);
          q += __shfl_xor(q, mk);
        }
        if (ln15 == 0) {
          int r = m * 16 + hi16 * 8 + v;     // block-local row
          part_s[r * NW + wave] = s;
          part_q[r * NW + wave] = q;
        }
      }
    }
    __syncthreads();
    if (threadIdx.x < 32) {
      float S = 0.f, Q = 0.f;
      for (int w = 0; w < NW; ++w) {
        S += part_s[threadIdx.x * NW + w];
        Q += part_q[threadIdx.x * NW + w];
      }
      float mu = S / (float)C;
      rmean[threadIdx.x] = mu;
      rrstd[threadIdx.x] = rsqrtf(Q / (float)C - mu * mu + 1e-5f);
    }
    __syncthreads();
  }

  if (MODE == 0) {                 // LayerNorm + mish
    float gam[4], bet[4];
#pragma unroll
    for (int t = 0; t < 4; ++t) {
      int c = base + t * 16 + ln15;
      gam[t] = gamma[c];
      bet[t] = beta[c];
    }
#pragma unroll
    for (int m = 0; m < 2; ++m)
#pragma unroll
      for (int t = 0; t < 4; ++t)
#pragma unroll
        for (int v = 0; v < 8; ++v) {
          int r = m * 16 + hi16 * 8 + v;
          int c = base + t * 16 + ln15;
          float x = (acc[m][t][v] - rmean[r]) * rrstd[r] * gam[t] + bet[t];
          float sp = fmaxf(x, 0.f) + log1pf(expf(-fabsf(x)));
          float y = x * tanhf(sp);
          Yh[(size_t)(row0 + r) * ldyh + c] = (_Float16)y;
        }
  } else if (MODE == 1) {          // LayerNorm + simnorm (softmax over 8-col groups)
    float gam[4], bet[4];
#pragma unroll
    for (int t = 0; t < 4; ++t) {
      int c = base + t * 16 + ln15;
      gam[t] = gamma[c];
      bet[t] = beta[c];
    }
#pragma unroll
    for (int m = 0; m < 2; ++m)
#pragma unroll
      for (int t = 0; t < 4; ++t)
#pragma unroll
        for (int v = 0; v < 8; ++v) {
          int r = m * 16 + hi16 * 8 + v;
          int c = base + t * 16 + ln15;
          float x = (acc[m][t][v] - rmean[r]) * rrstd[r] * gam[t] + bet[t];
          // group of 8 consecutive cols = 8 consecutive lanes (masks 1,2,4 stay in group)
          float mx = x;
#pragma unroll
          for (int mk = 1; mk <= 4; mk <<= 1) mx = fmaxf(mx, __shfl_xor(mx, mk));
          float e = expf(x - mx);
          float sum = e;
#pragma unroll
          for (int mk = 1; mk <= 4; mk <<= 1) sum += __shfl_xor(sum, mk);
          Yh[(size_t)(row0 + r) * ldyh + c] = (_Float16)(e / sum);
        }
  } else {                         // head: bias-only, fp32 out, cols < OUT
#pragma unroll
    for (int m = 0; m < 2; ++m)
#pragma unroll
      for (int t = 0; t < 4; ++t) {
        int c = base + t * 16 + ln15;
        if (c < OUT) {
#pragma unroll
          for (int v = 0; v < 8; ++v) {
            int r = m * 16 + hi16 * 8 + v;
            Yf[(size_t)(row0 + r) * ldyf + c] = acc[m][t][v];
          }
        }
      }
  }
}

// ---------------------------------------------------------------------------
// Small helper kernels
// ---------------------------------------------------------------------------
__global__ void k_convert(const float* __restrict__ W, _Float16* __restrict__ D,
                          int O, int K, int Opad, int Kpad) {
  int i = blockIdx.x * blockDim.x + threadIdx.x;
  int tot = Opad * Kpad;
  if (i >= tot) return;
  int o = i / Kpad, k = i % Kpad;
  float v = (o < O && k < K) ? W[(size_t)o * K + k] : 0.f;
  D[i] = (_Float16)v;
}

__global__ void __launch_bounds__(256)
k_encode(const float* __restrict__ obs,
         const float* W0, const float* b0, const float* g0, const float* be0,
         const float* W1, const float* b1, const float* g1, const float* be1,
         float* __restrict__ z0) {
  __shared__ float h0[256];
  __shared__ float h1[512];
  __shared__ float stat[2];
  int t = threadIdx.x;
  float a = b0[t];
  for (int k = 0; k < 223; ++k) a += W0[t * 223 + k] * obs[k];
  h0[t] = a;
  __syncthreads();
  if (t == 0) {
    float s = 0.f, q = 0.f;
    for (int i = 0; i < 256; ++i) { s += h0[i]; q += h0[i] * h0[i]; }
    float m = s / 256.f;
    stat[0] = m; stat[1] = rsqrtf(q / 256.f - m * m + 1e-5f);
  }
  __syncthreads();
  {
    float v = (h0[t] - stat[0]) * stat[1] * g0[t] + be0[t];
    float sp = fmaxf(v, 0.f) + log1pf(expf(-fabsf(v)));
    h0[t] = v * tanhf(sp);
  }
  __syncthreads();
  for (int o = t; o < 512; o += 256) {
    float s = b1[o];
    for (int k = 0; k < 256; ++k) s += W1[o * 256 + k] * h0[k];
    h1[o] = s;
  }
  __syncthreads();
  if (t == 0) {
    float s = 0.f, q = 0.f;
    for (int i = 0; i < 512; ++i) { s += h1[i]; q += h1[i] * h1[i]; }
    float m = s / 512.f;
    stat[0] = m; stat[1] = rsqrtf(q / 512.f - m * m + 1e-5f);
  }
  __syncthreads();
  if (t < 64) {
    float vals[8];
    float mx = -1e30f;
    for (int u = 0; u < 8; ++u) {
      float x = (h1[t * 8 + u] - stat[0]) * stat[1] * g1[t * 8 + u] + be1[t * 8 + u];
      vals[u] = x; mx = fmaxf(mx, x);
    }
    float s = 0.f;
    for (int u = 0; u < 8; ++u) { vals[u] = expf(vals[u] - mx); s += vals[u]; }
    for (int u = 0; u < 8; ++u) z0[t * 8 + u] = vals[u] / s;
  }
}

// broadcast z0 into X (M x 576 fp16), zero cols 512..575
__global__ void k_broadcast(const float* __restrict__ z0, _Float16* __restrict__ X, int M) {
  int i = blockIdx.x * blockDim.x + threadIdx.x;
  int tot = M * 576;
  if (i >= tot) return;
  int c = i % 576;
  X[i] = (_Float16)((c < 512) ? z0[c] : 0.f);
}

// write 38-dim actions into X cols 512..549, zero 550..575
__global__ void k_write_act(const float* __restrict__ act, _Float16* __restrict__ X, int M) {
  int i = blockIdx.x * blockDim.x + threadIdx.x;
  int tot = M * 64;
  if (i >= tot) return;
  int r = i / 64, c = i % 64;
  float v = (c < 38) ? act[(size_t)r * 38 + c] : 0.f;
  X[(size_t)r * 576 + 512 + c] = (_Float16)v;
}

__global__ void k_zero(float* __restrict__ p, int n) {
  int i = blockIdx.x * blockDim.x + threadIdx.x;
  if (i < n) p[i] = 0.f;
}

__global__ void k_init_meanstd(float* __restrict__ mean, float* __restrict__ stdv) {
  int i = blockIdx.x * blockDim.x + threadIdx.x;
  if (i < 190) { mean[i] = 0.f; stdv[i] = 2.0f; }
}

// tanh-Gaussian policy sample from head output (M x 76 fp32)
__global__ void k_pi_sample(const float* __restrict__ head, float* __restrict__ act_out,
                            int M, unsigned long long seed) {
  int i = blockIdx.x * blockDim.x + threadIdx.x;
  int tot = M * 38;
  if (i >= tot) return;
  int r = i / 38, d = i % 38;
  float mu = head[(size_t)r * 76 + d];
  float ls = head[(size_t)r * 76 + 38 + d];
  ls = -10.f + 0.5f * 12.f * (tanhf(ls) + 1.f);
  float eps = nrand(seed, (unsigned long long)i);
  act_out[(size_t)r * 38 + d] = tanhf(mu + eps * expf(ls));
}

// CEM gaussian actions, rows 256..8191 of each horizon slice
__global__ void k_gauss(const float* __restrict__ mean, const float* __restrict__ stdv,
                        float* __restrict__ actions, unsigned long long seed) {
  int i = blockIdx.x * blockDim.x + threadIdx.x;
  const int tot = 5 * 7936 * 38;
  if (i >= tot) return;
  int t = i / (7936 * 38);
  int rem = i % (7936 * 38);
  int r = 256 + rem / 38;
  int d = rem % 38;
  float v = mean[t * 38 + d] + stdv[t * 38 + d] * nrand(seed, (unsigned long long)i);
  v = fminf(1.f, fmaxf(-1.f, v));
  actions[((size_t)t * 8192 + r) * 38 + d] = v;
}

DEV_INLINE float two_hot_inv_row(const float* L) {
  float mx = -1e30f;
  for (int i = 0; i < 101; ++i) mx = fmaxf(mx, L[i]);
  float s = 0.f, ev = 0.f;
  for (int i = 0; i < 101; ++i) {
    float e = expf(L[i] - mx);
    s += e;
    ev += e * (-10.f + 0.2f * (float)i);
  }
  float v = ev / s;
  return copysignf(expm1f(fabsf(v)), v);
}

__global__ void k_two_hot_accum(const float* __restrict__ logits, float* __restrict__ G,
                                float disc, int M) {
  int r = blockIdx.x * blockDim.x + threadIdx.x;
  if (r >= M) return;
  G[r] += disc * two_hot_inv_row(logits + (size_t)r * 101);
}

__global__ void k_value(const float* __restrict__ qlog, const float* __restrict__ G,
                        float* __restrict__ value, float disc5, int M) {
  int r = blockIdx.x * blockDim.x + threadIdx.x;
  if (r >= M) return;
  float acc = 0.f;
  for (int q = 0; q < 5; ++q)
    acc += two_hot_inv_row(qlog + ((size_t)q * M + r) * 101);
  float val = G[r] + disc5 * (acc / 5.f);
  if (!isfinite(val)) val = 0.f;
  value[r] = val;
}

// single-block bitonic sort (descending) over 8192 values; emit top-512
__global__ void __launch_bounds__(1024)
k_topk(const float* __restrict__ value, int* __restrict__ elite_idx,
       float* __restrict__ elite_val) {
  __shared__ float sv[8192];
  __shared__ int   si[8192];
  for (int i = threadIdx.x; i < 8192; i += 1024) { sv[i] = value[i]; si[i] = i; }
  __syncthreads();
  for (int k = 2; k <= 8192; k <<= 1) {
    for (int j = k >> 1; j > 0; j >>= 1) {
      for (int i = threadIdx.x; i < 8192; i += 1024) {
        int ixj = i ^ j;
        if (ixj > i) {
          bool desc = ((i & k) == 0);
          bool sw = desc ? (sv[i] < sv[ixj]) : (sv[i] > sv[ixj]);
          if (sw) {
            float tv = sv[i]; sv[i] = sv[ixj]; sv[ixj] = tv;
            int ti = si[i]; si[i] = si[ixj]; si[ixj] = ti;
          }
        }
      }
      __syncthreads();
    }
  }
  for (int i = threadIdx.x; i < 512; i += 1024) {
    elite_idx[i] = si[i];
    elite_val[i] = sv[i];
  }
}

__global__ void __launch_bounds__(512)
k_score(const float* __restrict__ elite_val, float* __restrict__ score,
        float* __restrict__ ssum) {
  __shared__ float sh[512];
  int t = threadIdx.x;
  float vmax = elite_val[0];
  float s = expf(0.5f * (elite_val[t] - vmax));
  score[t] = s;
  sh[t] = s;
  __syncthreads();
  for (int o = 256; o > 0; o >>= 1) {
    if (t < o) sh[t] += sh[t + o];
    __syncthreads();
  }
  if (t == 0) ssum[0] = sh[0] + 1e-9f;
}

__global__ void k_meanstd(const float* __restrict__ score, const float* __restrict__ ssum,
                          const int* __restrict__ elite_idx, const float* __restrict__ actions,
                          float* __restrict__ mean, float* __restrict__ stdv) {
  int i = blockIdx.x * blockDim.x + threadIdx.x;
  if (i >= 190) return;
  int t = i / 38, d = i % 38;
  float S = ssum[0];
  float m = 0.f;
  for (int e = 0; e < 512; ++e)
    m += score[e] * actions[((size_t)t * 8192 + elite_idx[e]) * 38 + d];
  m /= S;
  float v = 0.f;
  for (int e = 0; e < 512; ++e) {
    float a = actions[((size_t)t * 8192 + elite_idx[e]) * 38 + d];
    v += score[e] * (a - m) * (a - m);
  }
  v /= S;
  float s = sqrtf(v);
  s = fminf(2.f, fmaxf(0.05f, s));
  mean[i] = m;
  stdv[i] = s;
}

__global__ void k_final(const float* __restrict__ score, const float* __restrict__ ssum,
                        const int* __restrict__ elite_idx, const float* __restrict__ actions,
                        float* __restrict__ out, unsigned long long seed) {
  if (threadIdx.x != 0 || blockIdx.x != 0) return;
  float u = u01(sm64(seed)) * ssum[0];
  float c = 0.f;
  int pick = 511;
  for (int e = 0; e < 512; ++e) {
    c += score[e];
    if (u <= c) { pick = e; break; }
  }
  int row = elite_idx[pick];
  for (int d = 0; d < 38; ++d) {
    float a = actions[(size_t)row * 38 + d];   // horizon t=0 slice
    out[d] = fminf(1.f, fmaxf(-1.f, a));
  }
}

// ---------------------------------------------------------------------------
// Host orchestration
// ---------------------------------------------------------------------------
static inline int cdiv(int a, int b) { return (a + b - 1) / b; }

extern "C" void kernel_launch(void* const* d_in, const int* in_sizes, int n_in,
                              void* d_out, int out_size, void* d_ws, size_t ws_size,
                              hipStream_t stream) {
  (void)in_sizes; (void)n_in; (void)out_size; (void)ws_size;

  // ---- input leaf order (dict keys sorted: dyn, enc, pi, qs, rew) ----
  auto F = [&](int i) { return (const float*)d_in[i]; };
  const float* obs = F(0);
  // dyn: 1..12  enc: 13..20  pi: 21..30  qs: 31..40  rew: 41..50

  // ---- workspace layout ----
  char* base = (char*)d_ws;
  size_t off = 0;
  auto A = [&](size_t bytes) -> char* {
    off = (off + 255) & ~(size_t)255;
    char* p = base + off;
    off += bytes;
    return p;
  };
  _Float16* Wdyn0 = (_Float16*)A((size_t)512 * 576 * 2);
  _Float16* Wdyn1 = (_Float16*)A((size_t)512 * 512 * 2);
  _Float16* Wdyn2 = (_Float16*)A((size_t)512 * 512 * 2);
  _Float16* Wrew0 = (_Float16*)A((size_t)512 * 576 * 2);
  _Float16* Wrew1 = (_Float16*)A((size_t)512 * 512 * 2);
  _Float16* Wrew2 = (_Float16*)A((size_t)128 * 512 * 2);
  _Float16* Wpi0  = (_Float16*)A((size_t)512 * 512 * 2);
  _Float16* Wpi1  = (_Float16*)A((size_t)512 * 512 * 2);
  _Float16* Wpi2  = (_Float16*)A((size_t)128 * 512 * 2);
  _Float16* Wq0   = (_Float16*)A((size_t)5 * 512 * 576 * 2);
  _Float16* Wq1   = (_Float16*)A((size_t)5 * 512 * 512 * 2);
  _Float16* Wq2   = (_Float16*)A((size_t)5 * 128 * 512 * 2);
  _Float16* Xa    = (_Float16*)A((size_t)8192 * 576 * 2);
  _Float16* Xb    = (_Float16*)A((size_t)8192 * 576 * 2);
  _Float16* H1    = (_Float16*)A((size_t)8192 * 512 * 2);
  _Float16* H2    = (_Float16*)A((size_t)8192 * 512 * 2);
  float* z0f   = (float*)A(512 * 4);
  float* RewL  = (float*)A((size_t)8192 * 101 * 4);
  float* PiH   = (float*)A((size_t)8192 * 76 * 4);
  float* QL    = (float*)A((size_t)5 * 8192 * 101 * 4);
  float* G     = (float*)A(8192 * 4);
  float* Val   = (float*)A(8192 * 4);
  float* Act   = (float*)A((size_t)6 * 8192 * 38 * 4);
  float* Mean  = (float*)A(190 * 4);
  float* Std   = (float*)A(190 * 4);
  float* Score = (float*)A(512 * 4);
  float* Ssum  = (float*)A(4);
  int*   EIdx  = (int*)A(512 * 4);
  float* EVal  = (float*)A(512 * 4);

  // ---- weight repack fp32 -> padded fp16 ----
  auto conv = [&](const float* src, _Float16* dst, int O, int K, int Op, int Kp) {
    k_convert<<<cdiv(Op * Kp, 256), 256, 0, stream>>>(src, dst, O, K, Op, Kp);
  };
  conv(F(1),  Wdyn0, 512, 550, 512, 576);
  conv(F(5),  Wdyn1, 512, 512, 512, 512);
  conv(F(9),  Wdyn2, 512, 512, 512, 512);
  conv(F(41), Wrew0, 512, 550, 512, 576);
  conv(F(45), Wrew1, 512, 512, 512, 512);
  conv(F(49), Wrew2, 101, 512, 128, 512);
  conv(F(21), Wpi0,  512, 512, 512, 512);
  conv(F(25), Wpi1,  512, 512, 512, 512);
  conv(F(29), Wpi2,   76, 512, 128, 512);
  for (int q = 0; q < 5; ++q) {
    conv(F(31) + (size_t)q * 512 * 550, Wq0 + (size_t)q * 512 * 576, 512, 550, 512, 576);
    conv(F(35) + (size_t)q * 512 * 512, Wq1 + (size_t)q * 512 * 512, 512, 512, 512, 512);
    conv(F(39) + (size_t)q * 101 * 512, Wq2 + (size_t)q * 128 * 512, 101, 512, 128, 512);
  }

  // ---- encoder (1 row, scalar) ----
  k_encode<<<1, 256, 0, stream>>>(obs, F(13), F(14), F(15), F(16),
                                  F(17), F(18), F(19), F(20), z0f);

  // ---- GEMM dispatch helper (32 rows per block) ----
  auto gemm = [&](int mode, const _Float16* X, int ldx, const _Float16* Wp, int ldw,
                  const float* bias, const float* gam, const float* bet,
                  _Float16* Yh, int ldyh, float* Yf, int ldyf, int M, int K, int OUT) {
    dim3 grid(M / 32);
    if (mode == 0)
      gemm_ln_act<8, 0><<<grid, 256, 0, stream>>>(X, ldx, Wp, ldw, bias, gam, bet,
                                                  Yh, ldyh, Yf, ldyf, K, OUT);
    else if (mode == 1)
      gemm_ln_act<8, 1><<<grid, 256, 0, stream>>>(X, ldx, Wp, ldw, bias, gam, bet,
                                                  Yh, ldyh, Yf, ldyf, K, OUT);
    else
      gemm_ln_act<2, 2><<<grid, 64, 0, stream>>>(X, ldx, Wp, ldw, bias, gam, bet,
                                                 Yh, ldyh, Yf, ldyf, K, OUT);
  };

  auto pi_chain = [&](int M, const _Float16* Xc) {
    gemm(0, Xc, 576, Wpi0, 512, F(22), F(23), F(24), H1, 512, nullptr, 0, M, 512, 512);
    gemm(0, H1, 512, Wpi1, 512, F(26), F(27), F(28), H2, 512, nullptr, 0, M, 512, 512);
    gemm(2, H2, 512, Wpi2, 512, F(30), nullptr, nullptr, nullptr, 0, PiH, 76, M, 512, 76);
  };
  auto dyn_chain = [&](int M, const _Float16* Xc, _Float16* Xn) {
    gemm(0, Xc, 576, Wdyn0, 576, F(2),  F(3),  F(4),  H1, 512, nullptr, 0, M, 576, 512);
    gemm(0, H1, 512, Wdyn1, 512, F(6),  F(7),  F(8),  H2, 512, nullptr, 0, M, 512, 512);
    gemm(1, H2, 512, Wdyn2, 512, F(10), F(11), F(12), Xn, 576, nullptr, 0, M, 512, 512);
  };
  auto rew_chain = [&](int M, const _Float16* Xc) {
    gemm(0, Xc, 576, Wrew0, 576, F(42), F(43), F(44), H1, 512, nullptr, 0, M, 576, 512);
    gemm(0, H1, 512, Wrew1, 512, F(46), F(47), F(48), H2, 512, nullptr, 0, M, 512, 512);
    gemm(2, H2, 512, Wrew2, 512, F(50), nullptr, nullptr, nullptr, 0, RewL, 101, M, 512, 101);
  };
  auto q_chain = [&](int M, const _Float16* Xc, int q) {
    gemm(0, Xc, 576, Wq0 + (size_t)q * 512 * 576, 576,
         F(32) + q * 512, F(33) + q * 512, F(34) + q * 512, H1, 512, nullptr, 0, M, 576, 512);
    gemm(0, H1, 512, Wq1 + (size_t)q * 512 * 512, 512,
         F(36) + q * 512, F(37) + q * 512, F(38) + q * 512, H2, 512, nullptr, 0, M, 512, 512);
    gemm(2, H2, 512, Wq2 + (size_t)q * 128 * 512, 512,
         F(40) + q * 101, nullptr, nullptr, nullptr, 0,
         QL + (size_t)q * 8192 * 101, 101, M, 512, 101);
  };

  // ---- policy rollout: pi_actions for rows 0..255 of each horizon slice ----
  k_broadcast<<<cdiv(256 * 576, 256), 256, 0, stream>>>(z0f, Xa, 256);
  {
    _Float16* Xc = Xa; _Float16* Xn = Xb;
    for (int t = 0; t < 5; ++t) {
      pi_chain(256, Xc);
      k_pi_sample<<<cdiv(256 * 38, 256), 256, 0, stream>>>(
          PiH, Act + (size_t)t * 8192 * 38, 256, 1000ull + t);
      if (t < 4) {
        k_write_act<<<cdiv(256 * 64, 256), 256, 0, stream>>>(
            Act + (size_t)t * 8192 * 38, Xc, 256);
        dyn_chain(256, Xc, Xn);
        _Float16* tmp = Xc; Xc = Xn; Xn = tmp;
      }
    }
  }

  k_init_meanstd<<<1, 256, 0, stream>>>(Mean, Std);

  // ---- CEM iterations ----
  float disc5 = 1.f;
  for (int t = 0; t < 5; ++t) disc5 *= 0.99f;
  for (int it = 0; it < 6; ++it) {
    k_gauss<<<cdiv(5 * 7936 * 38, 256), 256, 0, stream>>>(Mean, Std, Act, 2000ull + it);
    k_broadcast<<<cdiv(8192 * 576, 256), 256, 0, stream>>>(z0f, Xa, 8192);
    k_zero<<<32, 256, 0, stream>>>(G, 8192);
    _Float16* Xc = Xa; _Float16* Xn = Xb;
    float disc = 1.f;
    for (int t = 0; t < 5; ++t) {
      k_write_act<<<cdiv(8192 * 64, 256), 256, 0, stream>>>(
          Act + (size_t)t * 8192 * 38, Xc, 8192);
      rew_chain(8192, Xc);
      k_two_hot_accum<<<32, 256, 0, stream>>>(RewL, G, disc, 8192);
      dyn_chain(8192, Xc, Xn);
      _Float16* tmp = Xc; Xc = Xn; Xn = tmp;
      disc *= 0.99f;
    }
    pi_chain(8192, Xc);
    k_pi_sample<<<cdiv(8192 * 38, 256), 256, 0, stream>>>(
        PiH, Act + (size_t)5 * 8192 * 38, 8192, 3000ull + it);
    k_write_act<<<cdiv(8192 * 64, 256), 256, 0, stream>>>(
        Act + (size_t)5 * 8192 * 38, Xc, 8192);
    for (int q = 0; q < 5; ++q) q_chain(8192, Xc, q);
    k_value<<<32, 256, 0, stream>>>(QL, G, Val, disc5, 8192);
    k_topk<<<1, 1024, 0, stream>>>(Val, EIdx, EVal);
    k_score<<<1, 512, 0, stream>>>(EVal, Score, Ssum);
    k_meanstd<<<1, 256, 0, stream>>>(Score, Ssum, EIdx, Act, Mean, Std);
  }

  k_final<<<1, 1, 0, stream>>>(Score, Ssum, EIdx, Act, (float*)d_out, 4000ull);
}
